// GcnGatModel1_45406394253547
// MI455X (gfx1250) — compile-verified, matrix operand order checked by reading, not run on hardware
//
#include <hip/hip_runtime.h>
#include <math.h>

// ---------------- model constants ----------------
#define B_G   16
#define NPG_  512
#define EPG_  8192
#define N_N   (B_G * NPG_)     // 8192 nodes
#define E_E   (B_G * EPG_)     // 131072 edges (w/o self loops)
#define ET_   (E_E + N_N)      // edges incl. self loops
#define BSH_  256
#define OUT_  10
#define BN_SCALE_ 0.9999950000374997f   // 1/sqrt(1+1e-5)
#define BETA2_    0.04879016416943205f  // log(1.05)
#define ALPHA2_   0.5f

typedef __attribute__((ext_vector_type(16))) __bf16 v16bf;
typedef __attribute__((ext_vector_type(8)))  __bf16 v8bf;
typedef __attribute__((ext_vector_type(8)))  float  v8f;

union V16U { v16bf v; v8bf h[2]; };

// ---------------- tiny utility kernels ----------------
__global__ void fill_f32_k(float* __restrict__ p, float v, long n) {
  long i = (long)blockIdx.x * blockDim.x + threadIdx.x;
  if (i < n) p[i] = v;
}
__global__ void fill_u32_k(unsigned* __restrict__ p, unsigned v, long n) {
  long i = (long)blockIdx.x * blockDim.x + threadIdx.x;
  if (i < n) p[i] = v;
}
__global__ void cvt_bf16_k(const float* __restrict__ x, __bf16* __restrict__ y, long n) {
  long i = (long)blockIdx.x * blockDim.x + threadIdx.x;
  if (i < n) y[i] = (__bf16)x[i];
}
// W [K,O] f32 row-major -> Wt [O,K] bf16 row-major (contiguous-K B fragments)
__global__ void wt_cvt_k(const float* __restrict__ W, __bf16* __restrict__ Wt, int K, int O) {
  long i = (long)blockIdx.x * blockDim.x + threadIdx.x;
  if (i >= (long)K * O) return;
  int k = (int)(i / O), o = (int)(i % O);
  Wt[(size_t)o * K + k] = (__bf16)W[i];
}

// ---------------- WMMA GEMM: C[M,O] = A[M,K] @ W[K,O] (+bias) ----------------
// A: bf16 row-major [M,K]; Bt: bf16 row-major [O,K] (W transposed).
// One wave -> 16 x (16*TN) strip: one A fragment is reused across TN B tiles
// (TN WMMAs per K-step), cutting activation re-reads TN-fold.
template <int TN>
__global__ void __launch_bounds__(256)
wmma_gemm_k(const __bf16* __restrict__ A, const __bf16* __restrict__ Bt,
            const float* __restrict__ bias, float* __restrict__ C,
            int M, int K, int O) {
  int wave = threadIdx.x >> 5;
  int lane = threadIdx.x & 31;
  int wtilesN = (O >> 4) / TN;           // wave strips along N
  int tile = blockIdx.x * 8 + wave;
  int mt = tile / wtilesN;
  int nt = tile - mt * wtilesN;
  if (mt >= (M >> 4)) return;
  int m0 = mt << 4, n0 = (nt * TN) << 4;
  int r16 = lane & 15;
  int kb  = (lane >> 4) << 3;            // K-phase: 0 or 8
  const __bf16* ap = A  + (size_t)(m0 + r16) * K + kb;
  const __bf16* bp = Bt + (size_t)(n0 + r16) * K + kb;
  v8f acc[TN];
#pragma unroll
  for (int t = 0; t < TN; ++t) acc[t] = (v8f){};
  for (int k0 = 0; k0 < K; k0 += 32) {
    V16U a;
    a.h[0] = *(const v8bf*)(ap + k0);        // K = k0+kb .. +7
    a.h[1] = *(const v8bf*)(ap + k0 + 16);   // K = k0+kb+16 .. +23
#pragma unroll
    for (int t = 0; t < TN; ++t) {
      const __bf16* bpt = bp + (size_t)t * 16 * K;
      V16U b;
      b.h[0] = *(const v8bf*)(bpt + k0);
      b.h[1] = *(const v8bf*)(bpt + k0 + 16);
      acc[t] = __builtin_amdgcn_wmma_f32_16x16x32_bf16(
          false, a.v, false, b.v, (short)0, acc[t], false, false);
    }
  }
  // C layout: lane[3:0] = N col, lane[4] selects M base (+8), VGPR r = M row.
  int rowb = m0 + ((lane >> 4) << 3);
#pragma unroll
  for (int t = 0; t < TN; ++t) {
    int c = n0 + t * 16 + r16;
    float bv = bias ? bias[c] : 0.0f;
#pragma unroll
    for (int r = 0; r < 8; ++r)
      C[(size_t)(rowb + r) * O + c] = acc[t][r] + bv;
  }
}

// ---------------- graph preprocessing ----------------
__global__ void build_edges_k(const int* __restrict__ el, int* __restrict__ src,
                              int* __restrict__ dst) {
  long e = (long)blockIdx.x * blockDim.x + threadIdx.x;
  if (e >= E_E) return;
  int b = (int)(e / EPG_);
  src[e] = el[e] + b * NPG_;
  dst[e] = el[(long)E_E + e] + b * NPG_;
}
__global__ void deg_scatter_k(const int* __restrict__ dst, const float* __restrict__ w,
                              float* __restrict__ deg) {
  long e = (long)blockIdx.x * blockDim.x + threadIdx.x;
  if (e >= E_E) return;
  atomicAdd(&deg[dst[e]], w ? w[e] : 1.0f);
}
__global__ void make_dinv_k(float* __restrict__ deg, int n) {
  int i = blockIdx.x * blockDim.x + threadIdx.x;
  if (i >= n) return;
  float d = deg[i];
  deg[i] = d > 0.0f ? rsqrtf(d) : 0.0f;
}
__global__ void edge_norm_k(const int* __restrict__ src, const int* __restrict__ dst,
                            const float* __restrict__ w, const float* __restrict__ dinv,
                            float* __restrict__ nrm) {
  long e = (long)blockIdx.x * blockDim.x + threadIdx.x;
  if (e >= E_E) return;
  nrm[e] = dinv[src[e]] * (w ? w[e] : 1.0f) * dinv[dst[e]];
}
__global__ void self_norm_k(const float* __restrict__ dinv, float* __restrict__ s, int n) {
  int i = blockIdx.x * blockDim.x + threadIdx.x;
  if (i >= n) return;
  s[i] = dinv[i] * dinv[i];
}

// ---------------- GCN / GCN2 ----------------
// One float4 per thread: one global_load_b128 + 4 atomics; F multiple of 4.
__global__ void scatter_agg_k(const int* __restrict__ src, const int* __restrict__ dst,
                              const float* __restrict__ nrm, const float* __restrict__ t,
                              float* __restrict__ agg, int F) {
  int F4 = F >> 2;
  long id = (long)blockIdx.x * blockDim.x + threadIdx.x;
  if (id >= (long)E_E * F4) return;
  int e = (int)(id / F4), q = (int)(id % F4);
  float w = nrm[e];
  float4 v = *(const float4*)(t + (size_t)src[e] * F + q * 4);
  float* o = agg + (size_t)dst[e] * F + q * 4;
  atomicAdd(o + 0, w * v.x);
  atomicAdd(o + 1, w * v.y);
  atomicAdd(o + 2, w * v.z);
  atomicAdd(o + 3, w * v.w);
}
// out = relu(g * ((agg + selfn*t + bias) * BN_SCALE) + bt)
__global__ void gcn_fin_k(const float* __restrict__ agg, const float* __restrict__ t,
                          const float* __restrict__ selfn, const float* __restrict__ bias,
                          const float* __restrict__ g, const float* __restrict__ bt,
                          float* __restrict__ out, long n, int F) {
  long id = (long)blockIdx.x * blockDim.x + threadIdx.x;
  if (id >= n) return;
  long i = id / F; int f = (int)(id % F);
  float v = agg[id] + selfn[i] * t[id] + bias[f];
  v = g[f] * (v * BN_SCALE_) + bt[f];
  out[id] = fmaxf(v, 0.0f);
}
// h = (1-a)*(agg + selfn*x) + a*x0
__global__ void gcn2_h_k(const float* __restrict__ agg, const float* __restrict__ x,
                         const float* __restrict__ selfn, const float* __restrict__ x0,
                         float* __restrict__ h, long n, int F) {
  long id = (long)blockIdx.x * blockDim.x + threadIdx.x;
  if (id >= n) return;
  long i = id / F;
  h[id] = (1.0f - ALPHA2_) * (agg[id] + selfn[i] * x[id]) + ALPHA2_ * x0[id];
}
// out = relu(g * (((1-b2)h + b2*hW) * BN_SCALE) + bt)
__global__ void gcn2_fin_k(const float* __restrict__ h, const float* __restrict__ hw,
                           const float* __restrict__ g, const float* __restrict__ bt,
                           float* __restrict__ out, long n, int F) {
  long id = (long)blockIdx.x * blockDim.x + threadIdx.x;
  if (id >= n) return;
  int f = (int)(id % F);
  float v = (1.0f - BETA2_) * h[id] + BETA2_ * hw[id];
  v = g[f] * (v * BN_SCALE_) + bt[f];
  out[id] = fmaxf(v, 0.0f);
}
// out = relu(g * ((x + bias) * BN_SCALE) + bt)  (GAT epilogue)
__global__ void bias_bn_relu_k(const float* __restrict__ x, const float* __restrict__ bias,
                               const float* __restrict__ g, const float* __restrict__ bt,
                               float* __restrict__ out, long n, int F) {
  long id = (long)blockIdx.x * blockDim.x + threadIdx.x;
  if (id >= n) return;
  int f = (int)(id % F);
  float v = x[id] + bias[f];
  v = g[f] * (v * BN_SCALE_) + bt[f];
  out[id] = fmaxf(v, 0.0f);
}

// ---------------- GATv2 ----------------
__device__ __forceinline__ unsigned f2ord(float f) {
  unsigned u = __float_as_uint(f);
  return (u & 0x80000000u) ? ~u : (u | 0x80000000u);
}
__device__ __forceinline__ float ord2f(unsigned k) {
  return (k & 0x80000000u) ? __uint_as_float(k & 0x7fffffffu) : __uint_as_float(~k);
}
__global__ void gat_logit_k(const int* __restrict__ src, const int* __restrict__ dst,
                            const float* __restrict__ xl, const float* __restrict__ xr,
                            const float* __restrict__ att, float* __restrict__ logits,
                            unsigned* __restrict__ mkey, int H, int Fh) {
  long id = (long)blockIdx.x * blockDim.x + threadIdx.x;
  if (id >= (long)ET_ * H) return;
  int e = (int)(id / H), h = (int)(id % H);
  int s = e < E_E ? src[e] : (e - E_E);
  int d = e < E_E ? dst[e] : (e - E_E);
  const float* pl = xl + (size_t)s * H * Fh + h * Fh;
  const float* pr = xr + (size_t)d * H * Fh + h * Fh;
  const float* pa = att + h * Fh;
  float acc = 0.0f;
  for (int f = 0; f < Fh; ++f) {
    float gg = pl[f] + pr[f];
    gg = gg > 0.0f ? gg : 0.2f * gg;
    acc += gg * pa[f];
  }
  logits[id] = acc;
  atomicMax(&mkey[d * H + h], f2ord(acc));
}
__global__ void gat_exp_k(const int* __restrict__ src, const int* __restrict__ dst,
                          const float* __restrict__ logits, const unsigned* __restrict__ mkey,
                          float* __restrict__ exv, float* __restrict__ den, int H) {
  long id = (long)blockIdx.x * blockDim.x + threadIdx.x;
  if (id >= (long)ET_ * H) return;
  int e = (int)(id / H), h = (int)(id % H);
  int d = e < E_E ? dst[e] : (e - E_E);
  float ex = expf(logits[id] - ord2f(mkey[d * H + h]));
  exv[id] = ex;
  atomicAdd(&den[d * H + h], ex);
}
// One float4 per thread; 4 consecutive features always share a head (Fh >= 32).
__global__ void gat_scatter_k(const int* __restrict__ src, const int* __restrict__ dst,
                              const float* __restrict__ exv, const float* __restrict__ den,
                              const float* __restrict__ xl, float* __restrict__ out,
                              int H, int Fh) {
  int HF = H * Fh, HF4 = HF >> 2;
  long id = (long)blockIdx.x * blockDim.x + threadIdx.x;
  if (id >= (long)ET_ * HF4) return;
  int e = (int)(id / HF4), q = (int)(id % HF4);
  int r = q * 4;
  int h = r / Fh;
  int s = e < E_E ? src[e] : (e - E_E);
  int d = e < E_E ? dst[e] : (e - E_E);
  float alpha = exv[(size_t)e * H + h] / (den[d * H + h] + 1e-16f);
  float4 v = *(const float4*)(xl + (size_t)s * HF + r);
  float* o = out + (size_t)d * HF + r;
  atomicAdd(o + 0, alpha * v.x);
  atomicAdd(o + 1, alpha * v.y);
  atomicAdd(o + 2, alpha * v.z);
  atomicAdd(o + 3, alpha * v.w);
}

// ---------------- pooling (diff-pool collapses to per-graph column sum) ----------------
__global__ void graph_sum_k(const float* __restrict__ x, float* __restrict__ pooled) {
  int b = blockIdx.x, f = threadIdx.x;  // blockDim = 256
  const float* p = x + (size_t)b * NPG_ * BSH_ + f;
  float acc = 0.0f;
  for (int n = 0; n < NPG_; ++n) acc += p[(size_t)n * BSH_];
  pooled[b * BSH_ + f] = acc;
}
__global__ void out_lin_k(const float* __restrict__ pooled, const float* __restrict__ W,
                          const float* __restrict__ b, float* __restrict__ out) {
  int id = threadIdx.x;
  if (id >= B_G * OUT_) return;
  int bg = id / OUT_, o = id % OUT_;
  float acc = b[o];
  for (int f = 0; f < BSH_; ++f) acc += pooled[bg * BSH_ + f] * W[f * OUT_ + o];
  out[id] = acc;
}

// ---------------- host orchestration ----------------
extern "C" void kernel_launch(void* const* d_in, const int* in_sizes, int n_in,
                              void* d_out, int out_size, void* d_ws, size_t ws_size,
                              hipStream_t stream) {
  (void)in_sizes; (void)out_size; (void)ws_size;
  const float* x      = (const float*)d_in[0];
  const float* w_edge = (const float*)d_in[1];
  const int*   elocal = (const int*)d_in[n_in - 1];
  auto P = [&](int i) { return (const float*)d_in[i]; };

  // ---- workspace carve ----
  size_t off = 0;
  auto alloc = [&](size_t bytes) {
    void* p = (char*)d_ws + off;
    off += (bytes + 255) & ~(size_t)255;
    return p;
  };
  int*      src   = (int*)alloc((size_t)E_E * 4);
  int*      dst   = (int*)alloc((size_t)E_E * 4);
  float*    nrmW  = (float*)alloc((size_t)E_E * 4);
  float*    nrm1  = (float*)alloc((size_t)E_E * 4);
  float*    selfW = (float*)alloc((size_t)N_N * 4);
  float*    self1 = (float*)alloc((size_t)N_N * 4);
  float*    dinvW = (float*)alloc((size_t)N_N * 4);
  float*    dinv1 = (float*)alloc((size_t)N_N * 4);
  float*    bufA  = (float*)alloc((size_t)N_N * 256 * 4);
  float*    bufB  = (float*)alloc((size_t)N_N * 256 * 4);
  float*    bufT  = (float*)alloc((size_t)N_N * 256 * 4);
  float*    bufAg = (float*)alloc((size_t)N_N * 256 * 4);
  float*    bufH2 = (float*)alloc((size_t)N_N * 256 * 4);
  float*    bufXr = (float*)alloc((size_t)N_N * 256 * 4);
  __bf16*   bufBf = (__bf16*)alloc((size_t)N_N * 256 * 2);
  __bf16*   wtBf  = (__bf16*)alloc((size_t)256 * 256 * 2);
  float*    x1    = (float*)alloc((size_t)N_N * 256 * 4);
  float*    x2    = (float*)alloc((size_t)N_N * 128 * 4);
  float*    x3    = (float*)alloc((size_t)N_N * 64 * 4);
  float*    x4    = (float*)alloc((size_t)N_N * 32 * 4);
  float*    logits = (float*)alloc((size_t)ET_ * 2 * 4);
  float*    exv    = (float*)alloc((size_t)ET_ * 2 * 4);
  unsigned* mkey   = (unsigned*)alloc((size_t)N_N * 2 * 4);
  float*    den    = (float*)alloc((size_t)N_N * 2 * 4);
  float*    pooled = (float*)alloc((size_t)B_G * BSH_ * 4);

  auto L = [](long n) { return dim3((unsigned)((n + 255) / 256)); };

  // ---- graph norms (weighted + unweighted, self loops folded into deg init) ----
  build_edges_k<<<L(E_E), 256, 0, stream>>>(elocal, src, dst);
  fill_f32_k<<<L(N_N), 256, 0, stream>>>(dinvW, 1.0f, N_N);
  deg_scatter_k<<<L(E_E), 256, 0, stream>>>(dst, w_edge, dinvW);
  make_dinv_k<<<L(N_N), 256, 0, stream>>>(dinvW, N_N);
  edge_norm_k<<<L(E_E), 256, 0, stream>>>(src, dst, w_edge, dinvW, nrmW);
  self_norm_k<<<L(N_N), 256, 0, stream>>>(dinvW, selfW, N_N);
  fill_f32_k<<<L(N_N), 256, 0, stream>>>(dinv1, 1.0f, N_N);
  deg_scatter_k<<<L(E_E), 256, 0, stream>>>(dst, nullptr, dinv1);
  make_dinv_k<<<L(N_N), 256, 0, stream>>>(dinv1, N_N);
  edge_norm_k<<<L(E_E), 256, 0, stream>>>(src, dst, nullptr, dinv1, nrm1);
  self_norm_k<<<L(N_N), 256, 0, stream>>>(dinv1, self1, N_N);

  // ---- layer helpers ----
  auto gemm = [&](const float* Af, const float* Wf, const float* bias, float* Cf,
                  int K, int O) {
    cvt_bf16_k<<<L((long)N_N * K), 256, 0, stream>>>(Af, bufBf, (long)N_N * K);
    wt_cvt_k<<<L((long)K * O), 256, 0, stream>>>(Wf, wtBf, K, O);
    if ((O & 63) == 0) {
      int tiles = (N_N / 16) * (O / 64);
      wmma_gemm_k<4><<<dim3((tiles + 7) / 8), 256, 0, stream>>>(bufBf, wtBf, bias,
                                                                Cf, N_N, K, O);
    } else {  // O == 32
      int tiles = (N_N / 16) * (O / 32);
      wmma_gemm_k<2><<<dim3((tiles + 7) / 8), 256, 0, stream>>>(bufBf, wtBf, bias,
                                                                Cf, N_N, K, O);
    }
  };
  auto gcn = [&](const float* hin, float* hout, int K, int O,
                 const float* W, const float* bb, const float* g, const float* bt,
                 const float* nrm, const float* selfn) {
    gemm(hin, W, nullptr, bufT, K, O);
    long n = (long)N_N * O;
    fill_f32_k<<<L(n), 256, 0, stream>>>(bufAg, 0.0f, n);
    scatter_agg_k<<<L((long)E_E * (O >> 2)), 256, 0, stream>>>(src, dst, nrm, bufT,
                                                               bufAg, O);
    gcn_fin_k<<<L(n), 256, 0, stream>>>(bufAg, bufT, selfn, bb, g, bt, hout, n, O);
  };
  auto gcn2 = [&](const float* xin, const float* x0, float* hout, int F,
                  const float* W, const float* g, const float* bt) {
    long n = (long)N_N * F;
    fill_f32_k<<<L(n), 256, 0, stream>>>(bufAg, 0.0f, n);
    scatter_agg_k<<<L((long)E_E * (F >> 2)), 256, 0, stream>>>(src, dst, nrmW, xin,
                                                               bufAg, F);
    gcn2_h_k<<<L(n), 256, 0, stream>>>(bufAg, xin, selfW, x0, bufH2, n, F);
    gemm(bufH2, W, nullptr, bufT, F, F);
    gcn2_fin_k<<<L(n), 256, 0, stream>>>(bufH2, bufT, g, bt, hout, n, F);
  };
  auto gat = [&](const float* xin, float* hout, int Fin, int Fh, int pb,
                 const float* g, const float* bt) {
    const int H = 2, HF = 2 * Fh;
    gemm(xin, P(pb + 0), P(pb + 1), bufH2, Fin, HF);  // xl = x@Wl + bl
    gemm(xin, P(pb + 2), P(pb + 3), bufXr, Fin, HF);  // xr = x@Wr + br
    fill_u32_k<<<L((long)N_N * H), 256, 0, stream>>>(mkey, 0u, (long)N_N * H);
    gat_logit_k<<<L((long)ET_ * H), 256, 0, stream>>>(src, dst, bufH2, bufXr,
                                                      P(pb + 4), logits, mkey, H, Fh);
    fill_f32_k<<<L((long)N_N * H), 256, 0, stream>>>(den, 0.0f, (long)N_N * H);
    gat_exp_k<<<L((long)ET_ * H), 256, 0, stream>>>(src, dst, logits, mkey, exv, den, H);
    long n = (long)N_N * HF;
    fill_f32_k<<<L(n), 256, 0, stream>>>(bufAg, 0.0f, n);
    gat_scatter_k<<<L((long)ET_ * (HF >> 2)), 256, 0, stream>>>(src, dst, exv, den,
                                                                bufH2, bufAg, H, Fh);
    bias_bn_relu_k<<<L(n), 256, 0, stream>>>(bufAg, P(pb + 5), g, bt, hout, n, HF);
  };

  // ---- encoder: 12 GCN layers, skips after 2/5/8/11 ----
  const int encK[12] = {128, 256, 256, 256, 128, 128, 128, 64, 64, 64, 32, 32};
  const int encO[12] = {256, 256, 256, 128, 128, 128, 64, 64, 64, 32, 32, 32};
  float* encOut[12] = {bufA, bufB, x1, bufA, bufB, x2, bufA, bufB, x3, bufA, bufB, x4};
  const float* hin = x;
  for (int i = 0; i < 12; ++i) {
    int pi = 2 + 4 * i;
    gcn(hin, encOut[i], encK[i], encO[i], P(pi), P(pi + 1), P(pi + 2), P(pi + 3),
        nrmW, selfW);
    hin = encOut[i];
  }

  // ---- attention block ----
  gat(x4, bufA, 32, 32, 50, P(56), P(57));                              // -> 64
  gcn2(bufA, x3, bufB, 64, P(58), P(59), P(60));                        // -> 64
  gcn(bufB, bufA, 64, 64, P(61), P(62), P(63), P(64), nrmW, selfW);     // -> 64
  gat(bufA, bufB, 64, 64, 65, P(71), P(72));                            // -> 128

  // ---- decoder ----
  gcn2(bufB, x2, bufA, 128, P(73), P(74), P(75));                        // -> 128
  gcn(bufA, bufB, 128, 128, P(76), P(77), P(78), P(79), nrmW, selfW);    // -> 128
  gcn(bufB, bufA, 128, 256, P(80), P(81), P(82), P(83), nrm1, self1);    // -> 256 (unweighted)
  gcn2(bufA, x1, bufB, 256, P(84), P(85), P(86));                        // -> 256
  gcn(bufB, bufA, 256, 256, P(87), P(88), P(89), P(90), nrmW, selfW);    // -> 256
  gcn(bufA, bufB, 256, 256, P(91), P(92), P(93), P(94), nrmW, selfW);    // x_dec

  // ---- diff-pool collapses to per-graph column sums; then output linear ----
  graph_sum_k<<<dim3(B_G), 256, 0, stream>>>(bufB, pooled);
  out_lin_k<<<dim3(1), 256, 0, stream>>>(pooled, (const float*)d_in[n_in - 3],
                                         (const float*)d_in[n_in - 2], (float*)d_out);
}
